// RollingAttentionProbe_3599182594246
// MI455X (gfx1250) — matrix operand order repaired and checked by reading, not compiled, not run on hardware
//
#include <hip/hip_runtime.h>
#include <hip/hip_bf16.h>

// ---------------------------------------------------------------------------
// RollingAttentionProbe for MI455X (gfx1250, wave32, WMMA).
//
// Roofline: GEMM1 (32768x5376 @ 5376x100) = 35.2 GFLOP over 704 MB of x
// -> intensity ~50 FLOP/B -> HBM-bound (~30us @ 23.3 TB/s). bf16 WMMA with
// f32 accumulation makes compute negligible; read x once, stream through
// v_wmma_f32_16x16x32_bf16 with W1 staged ping-pong in LDS, fuse the full
// MLP + q/v probe in one kernel, then a light rolling-softmax pass.
// ---------------------------------------------------------------------------

typedef __attribute__((ext_vector_type(16))) __bf16 v16bf;
typedef __attribute__((ext_vector_type(8)))  __bf16 v8bf;
typedef __attribute__((ext_vector_type(8)))  float  v8f;

#define D_MODEL 5376
#define D_HID   100
#define NH      10
#define WIN     10
#define BB      8
#define TT      4096
#define NROW    (BB * TT)      // 32768 rows (b*T + t)
#define NCT     7              // 7 col tiles of 16 -> N padded to 112
#define KIT1    (D_MODEL / 32) // 168 K-steps for GEMM1
#define KIT2    4              // K padded to 128 for GEMM2 / probe
#define STG     (NCT * 32 * 8) // 1792 uints per W1 stage buffer

// ---- f32 pair -> packed bf16 ----------------------------------------------
#if __has_builtin(__builtin_amdgcn_cvt_pk_bf16_f32)
__device__ __forceinline__ unsigned pack2(float lo, float hi) {
  auto p = __builtin_amdgcn_cvt_pk_bf16_f32(lo, hi);  // v_cvt_pk_bf16_f32
  unsigned u; __builtin_memcpy(&u, &p, 4); return u;
}
#else
__device__ __forceinline__ unsigned pack2(float lo, float hi) { // manual RNE
  unsigned ul = __float_as_uint(lo), uh = __float_as_uint(hi);
  ul = (ul + 0x7FFFu + ((ul >> 16) & 1u)) >> 16;
  uh = (uh + 0x7FFFu + ((uh >> 16) & 1u)) & 0xFFFF0000u;
  return ul | uh;
}
#endif
__device__ __forceinline__ unsigned short f2bf(float f) {
  return (unsigned short)(pack2(f, 0.0f) & 0xFFFFu);
}

// ---------------------------------------------------------------------------
// Kernel 1: per 16-row tile (one wave): y = relu(relu(x W1^T + b1) W2^T + b2)
// then attn = y q^T, vals = y v^T.  8 waves/block -> 128 rows/block.
// LDS: sW2 = W2 staged in WMMA B-layout bf16 (28 KB, persistent)
//      sU  = union { W1 ping-pong stage (2 x 7 KB) ,
//                    per-wave h1/y bf16 row-major 16x128 (32 KB) }
// ---------------------------------------------------------------------------
__global__ __launch_bounds__(256) void fused_mlp_probe(
    const float* __restrict__ x,  const float* __restrict__ W1,
    const float* __restrict__ b1, const float* __restrict__ W2,
    const float* __restrict__ b2, const float* __restrict__ qm,
    const float* __restrict__ vm, float* __restrict__ attnW,
    float* __restrict__ valsW)
{
  __shared__ unsigned sW2[NCT * KIT2 * 32 * 8]; // 28672 B
  __shared__ unsigned sU[8 * 16 * 64];          // 32768 B

  const int tid  = threadIdx.x;
  const int lane = tid & 31;
  const int wave = tid >> 5;
  const int hi   = (lane >> 4) & 1;   // lane half selects K sub-block
  const int nlo  = lane & 15;         // column / row-in-tile index
  const int rowBase = blockIdx.x * 128 + wave * 16;

  // ---- Stage W2 into B-layout bf16 (B[k][n] = W2[n*100+k], K,N padded) ----
  for (int t = tid; t < NCT * KIT2 * 32; t += 256) {
    int l  = t & 31;
    int ct = t >> 7;
    int kb = ((t >> 5) & 3) * 32 + ((l >> 4) & 1) * 16;
    int n  = ct * 16 + (l & 15);
    float f[16];
#pragma unroll
    for (int i = 0; i < 16; ++i) {
      int k = kb + i;
      f[i] = (n < D_HID && k < D_HID) ? W2[n * D_HID + k] : 0.0f;
    }
    unsigned* d2 = &sW2[(unsigned)t * 8];
#pragma unroll
    for (int j = 0; j < 8; ++j) d2[j] = pack2(f[2 * j], f[2 * j + 1]);
  }

  // ---- W1 stage helper (as lambda): chunk kb -> buffer buf ----------------
  auto stageW1 = [&](int buf, int kb) {
    if (tid < NCT * 32) {
      int l  = tid & 31;
      int n  = (tid >> 5) * 16 + (l & 15);
      int ko = kb + ((l >> 4) & 1) * 16;
      unsigned* dst = &sU[(unsigned)(buf * STG) + (unsigned)tid * 8];
      if (n < D_HID) {
        const float4* src = reinterpret_cast<const float4*>(W1 + (size_t)n * D_MODEL + ko);
#pragma unroll
        for (int j = 0; j < 4; ++j) {
          float4 f = src[j];
          dst[2 * j]     = pack2(f.x, f.y);
          dst[2 * j + 1] = pack2(f.z, f.w);
        }
      } else {
#pragma unroll
        for (int j = 0; j < 8; ++j) dst[j] = 0u;
      }
    }
  };

  // ---- GEMM1: acc[ct] += A(16x32 bf16 of x) * B(32x16 bf16 of W1) ---------
  v8f acc[NCT];
#pragma unroll
  for (int c = 0; c < NCT; ++c) acc[c] = (v8f){0,0,0,0,0,0,0,0};

  const float* xrow = x + (size_t)(rowBase + nlo) * D_MODEL;

  stageW1(0, 0);
  __syncthreads();

  for (int it = 0; it < KIT1; ++it) {
    const int kb  = it * 32;
    const int cur = it & 1;
    if (it + 1 < KIT1) stageW1(cur ^ 1, kb + 32);   // overlap next-chunk stage

    // A fragment: contiguous K order -> lane gets 16 consecutive floats at kb + 16*hi
    const float4* pa0 = reinterpret_cast<const float4*>(xrow + kb + 16 * hi);
    const float4* pa1 = reinterpret_cast<const float4*>(xrow + kb + 16 * hi + 8);
    float4 f0 = pa0[0], f1 = pa0[1], f2 = pa1[0], f3 = pa1[1];
    if (it + 4 < KIT1) __builtin_prefetch(xrow + kb + 128, 0, 1); // global_prefetch_b8
    unsigned aw[8];
    aw[0] = pack2(f0.x, f0.y); aw[1] = pack2(f0.z, f0.w);
    aw[2] = pack2(f1.x, f1.y); aw[3] = pack2(f1.z, f1.w);
    aw[4] = pack2(f2.x, f2.y); aw[5] = pack2(f2.z, f2.w);
    aw[6] = pack2(f3.x, f3.y); aw[7] = pack2(f3.z, f3.w);
    v16bf a; __builtin_memcpy(&a, aw, 32);

    // Batch-load all 7 B fragments, then 7 back-to-back WMMAs (no RAW hazard)
    v16bf bm[NCT];
#pragma unroll
    for (int c = 0; c < NCT; ++c)
      bm[c] = *reinterpret_cast<const v16bf*>(&sU[(unsigned)(cur * STG) + (unsigned)(c * 32 + lane) * 8]);
#pragma unroll
    for (int c = 0; c < NCT; ++c)
      acc[c] = __builtin_amdgcn_wmma_f32_16x16x32_bf16(false, a, false, bm[c],
                                                       (short)0, acc[c], false, false);
    __syncthreads();   // reads of buf[cur] done; writes of buf[cur^1] visible
  }

  // ---- Epilogue 1: h1 = relu(acc + b1) -> LDS bf16 row-major 16x128 -------
  unsigned short* sY = reinterpret_cast<unsigned short*>(sU);
#pragma unroll
  for (int c = 0; c < NCT; ++c) {
    int n = c * 16 + nlo;
    float bb1 = (n < D_HID) ? b1[n] : 0.0f;
#pragma unroll
    for (int r = 0; r < 8; ++r) {
      int m = r + 8 * hi;                       // C/D layout: M = r + 8*laneHi
      float h1 = (n < D_HID) ? fmaxf(acc[c][r] + bb1, 0.0f) : 0.0f;
      sY[(wave * 16 + m) * 128 + n] = f2bf(h1);
    }
  }
#pragma unroll
  for (int i = 0; i < 8; ++i) {                 // zero K = 112..127 padding
    int idx = lane + 32 * i;
    sY[(wave * 16 + (idx >> 4)) * 128 + 112 + (idx & 15)] = 0;
  }
  __syncthreads();

  // ---- GEMM2: y = relu(h1 W2^T + b2), K=128, N padded to 112 --------------
  v8f acc2[NCT];
#pragma unroll
  for (int c = 0; c < NCT; ++c) acc2[c] = (v8f){0,0,0,0,0,0,0,0};
#pragma unroll
  for (int ki = 0; ki < KIT2; ++ki) {
    int kbase = ki * 32 + 16 * hi;
    const unsigned short* rp = &sY[(wave * 16 + nlo) * 128 + kbase];
    v8bf alo = *reinterpret_cast<const v8bf*>(rp);
    v8bf ahi = *reinterpret_cast<const v8bf*>(rp + 8);
    v16bf a;
#pragma unroll
    for (int i = 0; i < 8; ++i) { a[i] = alo[i]; a[i + 8] = ahi[i]; }
    v16bf bm[NCT];
#pragma unroll
    for (int c = 0; c < NCT; ++c)
      bm[c] = *reinterpret_cast<const v16bf*>(&sW2[(unsigned)((c * 4 + ki) * 32 + lane) * 8]);
#pragma unroll
    for (int c = 0; c < NCT; ++c)
      acc2[c] = __builtin_amdgcn_wmma_f32_16x16x32_bf16(false, a, false, bm[c],
                                                        (short)0, acc2[c], false, false);
  }

  // ---- Epilogue 2: y -> LDS bf16 (overwrite h1) ---------------------------
  __syncthreads();
#pragma unroll
  for (int c = 0; c < NCT; ++c) {
    int n = c * 16 + nlo;
    float bb2 = (n < D_HID) ? b2[n] : 0.0f;
#pragma unroll
    for (int r = 0; r < 8; ++r) {
      int m = r + 8 * hi;
      float yv = (n < D_HID) ? fmaxf(acc2[c][r] + bb2, 0.0f) : 0.0f;
      sY[(wave * 16 + m) * 128 + n] = f2bf(yv);
    }
  }
  __syncthreads();

  // ---- Probe: attn = y q^T, vals = y v^T (heads in cols 0..9) -------------
  v8f accA = (v8f){0,0,0,0,0,0,0,0};
  v8f accV = (v8f){0,0,0,0,0,0,0,0};
#pragma unroll
  for (int ki = 0; ki < KIT2; ++ki) {
    int kbase = ki * 32 + 16 * hi;
    const unsigned short* rp = &sY[(wave * 16 + nlo) * 128 + kbase];
    v8bf alo = *reinterpret_cast<const v8bf*>(rp);
    v8bf ahi = *reinterpret_cast<const v8bf*>(rp + 8);
    v16bf a;
#pragma unroll
    for (int i = 0; i < 8; ++i) { a[i] = alo[i]; a[i + 8] = ahi[i]; }
    unsigned qw[8], vw[8];
#pragma unroll
    for (int j = 0; j < 8; ++j) {
      int k0 = ki * 32 + 16 * hi + 2 * j;
      bool ok0 = (nlo < NH) && (k0 < D_HID);
      bool ok1 = (nlo < NH) && (k0 + 1 < D_HID);
      float q0 = ok0 ? qm[nlo * D_HID + k0] : 0.0f;
      float q1 = ok1 ? qm[nlo * D_HID + k0 + 1] : 0.0f;
      float v0 = ok0 ? vm[nlo * D_HID + k0] : 0.0f;
      float v1 = ok1 ? vm[nlo * D_HID + k0 + 1] : 0.0f;
      qw[j] = pack2(q0, q1); vw[j] = pack2(v0, v1);
    }
    v16bf bq, bv;
    __builtin_memcpy(&bq, qw, 32); __builtin_memcpy(&bv, vw, 32);
    accA = __builtin_amdgcn_wmma_f32_16x16x32_bf16(false, a, false, bq, (short)0, accA, false, false);
    accV = __builtin_amdgcn_wmma_f32_16x16x32_bf16(false, a, false, bv, (short)0, accV, false, false);
  }
#pragma unroll
  for (int r = 0; r < 8; ++r) {
    int row = rowBase + r + 8 * hi;
    attnW[(size_t)row * 16 + nlo] = accA[r];
    valsW[(size_t)row * 16 + nlo] = accV[r];
  }
}

// ---------------------------------------------------------------------------
// Kernel 2: rolling window-10 masked softmax pool + ordered-uint atomicMax.
// Float total order encode: pos -> bits|0x80000000, neg -> ~bits.
// ---------------------------------------------------------------------------
__global__ __launch_bounds__(256) void rolling_kernel(
    const float* __restrict__ attnW, const float* __restrict__ valsW,
    const unsigned char* __restrict__ mask, unsigned* __restrict__ maxbuf)
{
  int idx = blockIdx.x * 256 + threadIdx.x;
  if (idx >= NROW) return;
  int b = idx >> 12;
  int t = idx & (TT - 1);

  bool valid[WIN];
#pragma unroll
  for (int j = 0; j < WIN; ++j) {
    int p = t - (WIN - 1) + j;
    valid[j] = (p >= 0) && (mask[b * TT + p] != 0);
  }
  bool cur = mask[b * TT + t] != 0;

  for (int h = 0; h < NH; ++h) {
    float wa[WIN], wv[WIN];
    float mx = -3.4e38f; bool any = false;
#pragma unroll
    for (int j = 0; j < WIN; ++j) {
      if (valid[j]) {
        int p = t - (WIN - 1) + j;
        wa[j] = attnW[(size_t)(b * TT + p) * 16 + h];
        wv[j] = valsW[(size_t)(b * TT + p) * 16 + h];
        mx = fmaxf(mx, wa[j]); any = true;
      }
    }
    float res = 0.0f;
    if (any) {
      float s = 0.0f, num = 0.0f;
#pragma unroll
      for (int j = 0; j < WIN; ++j) {
        if (valid[j]) {
          float e = __expf(wa[j] - mx);
          s += e; num += e * wv[j];
        }
      }
      res = num / s;
    }
    if (!cur) res = -__builtin_inff();
    unsigned u = __float_as_uint(res);
    unsigned enc = (u & 0x80000000u) ? ~u : (u | 0x80000000u);
    atomicMax(&maxbuf[b * 16 + h], enc);   // GLOBAL_ATOMIC_MAX_U32
  }
}

__global__ void init_maxbuf(unsigned* maxbuf) {
  int i = threadIdx.x;
  if (i < BB * 16) maxbuf[i] = 0u;        // below every encoded float we emit
}

__global__ void finalize_kernel(const unsigned* __restrict__ maxbuf,
                                const float* __restrict__ bias,
                                float* __restrict__ out)
{
  int b = threadIdx.x;
  if (b < BB) {
    float s = 0.0f;
    for (int h = 0; h < NH; ++h) {
      unsigned u = maxbuf[b * 16 + h];
      unsigned bits = (u & 0x80000000u) ? (u & 0x7FFFFFFFu) : ~u;
      s += __uint_as_float(bits);
    }
    out[b] = s + bias[0];
  }
}

// ---------------------------------------------------------------------------
extern "C" void kernel_launch(void* const* d_in, const int* in_sizes, int n_in,
                              void* d_out, int out_size, void* d_ws, size_t ws_size,
                              hipStream_t stream) {
  (void)in_sizes; (void)n_in; (void)out_size; (void)ws_size;
  const float*         x    = (const float*)d_in[0];
  const unsigned char* mask = (const unsigned char*)d_in[1];
  const float*         W1   = (const float*)d_in[2];
  const float*         b1   = (const float*)d_in[3];
  const float*         W2   = (const float*)d_in[4];
  const float*         b2   = (const float*)d_in[5];
  const float*         qm   = (const float*)d_in[6];
  const float*         vm   = (const float*)d_in[7];
  const float*         bias = (const float*)d_in[8];
  float* out = (float*)d_out;

  float*    attnW  = (float*)d_ws;                       // 32768*16 f32 = 2 MB
  float*    valsW  = attnW + (size_t)NROW * 16;          // 2 MB
  unsigned* maxbuf = (unsigned*)(valsW + (size_t)NROW * 16);

  init_maxbuf<<<1, 128, 0, stream>>>(maxbuf);
  fused_mlp_probe<<<NROW / 128, 256, 0, stream>>>(x, W1, b1, W2, b2, qm, vm, attnW, valsW);
  rolling_kernel<<<NROW / 256, 256, 0, stream>>>(attnW, valsW, mask, maxbuf);
  finalize_kernel<<<1, 32, 0, stream>>>(maxbuf, bias, out);
}
